// Net_33767032881629
// MI455X (gfx1250) — matrix operand order repaired and checked by reading, not compiled, not run on hardware
//
#include <hip/hip_runtime.h>
#include <stdint.h>

#define BATCH   (1u << 20)
#define HID     64
#define DIN     4
#define DOUT    3
#define NLAYERS 4
#define BN_EPS  1e-5f

typedef __attribute__((ext_vector_type(16))) __bf16 v16bf;
typedef __attribute__((ext_vector_type(8)))  float  v8f;

// ---------- bf16 helpers (bit-level, RNE) ----------
__device__ __forceinline__ float bf2f(uint16_t u) {
    union { uint32_t u32; float f; } c; c.u32 = ((uint32_t)u) << 16; return c.f;
}
__device__ __forceinline__ uint16_t f2bf(float f) {
    union { float f; uint32_t u; } c; c.f = f;
    uint32_t u = c.u;
    uint32_t r = (u + 0x7FFFu + ((u >> 16) & 1u)) >> 16;
    return (uint16_t)r;
}
__device__ __forceinline__ void set_pair(v16bf& v, int e, uint32_t d) {
    union { uint32_t u; uint16_t s[2]; } c; c.u = d;
    union { uint16_t s; __bf16 b; } lo, hi; lo.s = c.s[0]; hi.s = c.s[1];
    v[e] = lo.b; v[e + 1] = hi.b;
}

// branch-free tanh: sign(x) * (1 - e)/(1 + e), e = exp(-2|x|)  (v_exp_f32 + v_rcp_f32)
__device__ __forceinline__ float fast_tanh(float x) {
    float a = fabsf(x);
    float e = __expf(-2.0f * a);
    float t = (1.0f - e) * __builtin_amdgcn_rcpf(1.0f + e);
    return copysignf(t, x);
}

// BN affine (+ optional residual) on one dword = 2 bf16
__device__ __forceinline__ uint32_t nr_dword(uint32_t zd, uint32_t yd,
                                             float sc0, float sh0, float sc1, float sh1,
                                             int resid) {
    float zl = bf2f((uint16_t)(zd & 0xffff));
    float zh = bf2f((uint16_t)(zd >> 16));
    float yl = resid ? bf2f((uint16_t)(yd & 0xffff)) : 0.f;
    float yh = resid ? bf2f((uint16_t)(yd >> 16)) : 0.f;
    float ol = sc0 * zl + sh0 + yl;
    float oh = sc1 * zh + sh1 + yh;
    return (uint32_t)f2bf(ol) | ((uint32_t)f2bf(oh) << 16);
}

// ---------- convert fp32 weights -> bf16, TRANSPOSED per matrix: out[m][n][k] ----------
__global__ __launch_bounds__(256)
void conv_w_bf16_t(const float* __restrict__ w, uint16_t* __restrict__ o, int nmat) {
    int ntot = nmat << 12;
    for (int i = blockIdx.x * blockDim.x + threadIdx.x; i < ntot; i += gridDim.x * blockDim.x) {
        int m = i >> 12, r = i & 4095;
        int k = r >> 6, n = r & 63;
        o[(m << 12) + n * HID + k] = f2bf(w[i]);
    }
}

// ---------- BN0: per-column sum / sumsq over x (1M x 4 fp32) ----------
__global__ __launch_bounds__(256)
void bn0_stats(const float4* __restrict__ x, float* __restrict__ gsum,
               float* __restrict__ gsq, int nrows) {
    __shared__ float ls[8];
    if (threadIdx.x < 8) ls[threadIdx.x] = 0.f;
    __syncthreads();
    float s[4] = {0, 0, 0, 0}, q[4] = {0, 0, 0, 0};
    const int stride = gridDim.x * blockDim.x;
    for (int r = blockIdx.x * blockDim.x + threadIdx.x; r < nrows; r += stride) {
        __builtin_prefetch(x + r + stride, 0, 0);
        float4 v = x[r];
        s[0] += v.x; q[0] += v.x * v.x;
        s[1] += v.y; q[1] += v.y * v.y;
        s[2] += v.z; q[2] += v.z * v.z;
        s[3] += v.w; q[3] += v.w * v.w;
    }
#pragma unroll
    for (int i = 0; i < 4; ++i) { atomicAdd(&ls[i], s[i]); atomicAdd(&ls[4 + i], q[i]); }
    __syncthreads();
    if (threadIdx.x < 4) {
        atomicAdd(&gsum[threadIdx.x], ls[threadIdx.x]);
        atomicAdd(&gsq[threadIdx.x],  ls[4 + threadIdx.x]);
    }
}

// ---------- BN finalize ----------
__global__ void bn_finalize(const float* __restrict__ gsum, const float* __restrict__ gsq,
                            const float* __restrict__ gamma, const float* __restrict__ beta,
                            float* __restrict__ scale, float* __restrict__ shift,
                            int dim, float inv_n) {
    int t = threadIdx.x;
    if (t < dim) {
        float mu  = gsum[t] * inv_n;
        float var = gsq[t] * inv_n - mu * mu;
        float inv = rsqrtf(var + BN_EPS);
        float sc  = gamma[t] * inv;
        scale[t] = sc;
        shift[t] = beta[t] - mu * sc;
    }
}

// ---------- layer 0: z = tanh(bn0(x) @ W0 + b0), bf16 out, accumulate stats ----------
__global__ __launch_bounds__(256)
void layer0_kernel(const float4* __restrict__ x,
                   const float* __restrict__ sc0, const float* __restrict__ sh0,
                   const float* __restrict__ W0, const float* __restrict__ b0,
                   uint16_t* __restrict__ zout,
                   float* __restrict__ gsum, float* __restrict__ gsq) {
    __shared__ float wl[DIN * HID];
    __shared__ float bl[HID];
    __shared__ float lsum[HID], lsq[HID];
    const int tid = threadIdx.x;
    if (tid < DIN * HID) wl[tid] = W0[tid];
    if (tid < HID) { bl[tid] = b0[tid]; lsum[tid] = 0.f; lsq[tid] = 0.f; }
    __syncthreads();
    const int row = blockIdx.x * 256 + tid;
    float4 xv = x[row];
    float h0 = sc0[0] * xv.x + sh0[0];
    float h1 = sc0[1] * xv.y + sh0[1];
    float h2 = sc0[2] * xv.z + sh0[2];
    float h3 = sc0[3] * xv.w + sh0[3];
    uint16_t* zr = zout + (size_t)row * HID;
#pragma unroll 8
    for (int j = 0; j < HID; ++j) {
        float a = bl[j] + h0 * wl[j] + h1 * wl[HID + j] + h2 * wl[2 * HID + j] + h3 * wl[3 * HID + j];
        float zz = fast_tanh(a);
        zr[j] = f2bf(zz);
        atomicAdd(&lsum[j], zz);
        atomicAdd(&lsq[j], zz * zz);
    }
    __syncthreads();
    if (tid < HID) { atomicAdd(&gsum[tid], lsum[tid]); atomicAdd(&gsq[tid], lsq[tid]); }
}

// ---------- fused WMMA GEMM layer ----------
// Fuses the PREVIOUS layer's BN affine (+residual) into A staging:
//   y = scp*z + shp (+ y_prev)   -> written back to ybuf AND used as A
//   z_new = tanh(y @ W + b)      -> zbuf (in place), + BN stats
// 256 threads = 8 waves; block covers 128 rows; wave computes 16 rows x 64 cols.
__global__ __launch_bounds__(256)
void gemm64_fused(uint16_t* zbuf,               // z in/out (bf16), same rows, no alias issue
                  uint16_t* ybuf,               // y in/out (bf16), in place
                  const float* __restrict__ scp, const float* __restrict__ shp,
                  const uint16_t* __restrict__ wtbf,  // bf16 W transposed [n][k]
                  const float* __restrict__ bias,
                  float* __restrict__ gsum, float* __restrict__ gsq,
                  int resid) {
    __shared__ __align__(16) uint16_t alds[128 * HID];   // 16 KB staged y rows
    __shared__ __align__(16) uint16_t wlds[HID * HID];   // 8 KB Wt[n][k]
    __shared__ float scl[HID], shl[HID], bl[HID];
    __shared__ float lsum[HID], lsq[HID];
    const int tid = threadIdx.x;

    if (tid < HID) {
        scl[tid] = scp[tid]; shl[tid] = shp[tid]; bl[tid] = bias[tid];
        lsum[tid] = 0.f; lsq[tid] = 0.f;
    }
    // W staging is a straight copy (pre-transposed on device)
    for (int t = tid; t < HID * HID; t += 256) wlds[t] = wtbf[t];
    __syncthreads();   // scl/shl ready for the fused transform below

    // stage A = BN-affine(z) (+ residual), write through to global y
    const size_t base = (size_t)blockIdx.x * (128 * HID);
    const uint4* zsrc = (const uint4*)(zbuf + base);
    const uint4* ysrc = (const uint4*)(ybuf + base);
    uint4* ydst = (uint4*)(ybuf + base);
    uint4* dst  = (uint4*)alds;
#pragma unroll
    for (int t = 0; t < 4; ++t) {
        const int i = tid + 256 * t;
        uint4 zv = zsrc[i];
        uint4 yv = {0, 0, 0, 0};
        if (resid) yv = ysrc[i];
        const int fb = (i * 8) & 63;           // 8 bf16 per uint4, HID==64
        uint4 ov;
        ov.x = nr_dword(zv.x, yv.x, scl[fb + 0], shl[fb + 0], scl[fb + 1], shl[fb + 1], resid);
        ov.y = nr_dword(zv.y, yv.y, scl[fb + 2], shl[fb + 2], scl[fb + 3], shl[fb + 3], resid);
        ov.z = nr_dword(zv.z, yv.z, scl[fb + 4], shl[fb + 4], scl[fb + 5], shl[fb + 5], resid);
        ov.w = nr_dword(zv.w, yv.w, scl[fb + 6], shl[fb + 6], scl[fb + 7], shl[fb + 7], resid);
        dst[i]  = ov;
        ydst[i] = ov;
    }
    __syncthreads();

    const int lane = tid & 31;
    const int wv   = tid >> 5;
    const int nl   = lane & 15;
    const int half = lane >> 4;
    const int r0l  = wv * 16;

    v8f acc[4];
#pragma unroll
    for (int nt = 0; nt < 4; ++nt)
#pragma unroll
        for (int v = 0; v < 8; ++v) acc[nt][v] = 0.f;

    // A fragment: lane m = nl, K split by half-wave (ISA 16-bit A 16x32 layout)
    const uint16_t* arow = &alds[(r0l + nl) * HID];
#pragma unroll
    for (int kb = 0; kb < 2; ++kb) {
        const int k0 = kb * 32;
        v16bf a;
#pragma unroll
        for (int p = 0; p < 8; ++p) {
            int k = k0 + ((p < 4) ? (8 * half + 2 * p) : (16 + 8 * half + 2 * (p - 4)));
            set_pair(a, 2 * p, *(const uint32_t*)&arow[k]);
        }
#pragma unroll
        for (int nt = 0; nt < 4; ++nt) {
            // B fragment: lane n = 16*nt+nl, k = k0 + 8*half + e (e<8), k0 + 16 + 8*half + (e-8) (e>=8) in Wt[n][k]
            const uint16_t* brow = &wlds[(16 * nt + nl) * HID + k0 + 8 * half];
            v16bf b;
#pragma unroll
            for (int p = 0; p < 8; ++p)
                set_pair(b, 2 * p, *(const uint32_t*)&brow[(p < 4) ? (2 * p) : (16 + 2 * (p - 4))]);
            acc[nt] = __builtin_amdgcn_wmma_f32_16x16x32_bf16(
                false, a, false, b, (short)0, acc[nt], false, false);
        }
    }

    // epilogue: bias + fast_tanh, store z bf16, reduce per-feature sum / sumsq
    const size_t rowbase = (size_t)blockIdx.x * 128 + r0l + 8 * half;
#pragma unroll
    for (int nt = 0; nt < 4; ++nt) {
        const int col = 16 * nt + nl;
        const float bb = bl[col];
        float s = 0.f, s2 = 0.f;
#pragma unroll
        for (int v = 0; v < 8; ++v) {
            float zz = fast_tanh(acc[nt][v] + bb);
            zbuf[(rowbase + v) * HID + col] = f2bf(zz);
            s += zz; s2 += zz * zz;
        }
        // lanes l and l^16 hold the same output column (m halves 0-7 / 8-15)
        s  += __shfl_xor(s, 16, 32);
        s2 += __shfl_xor(s2, 16, 32);
        if (half == 0) { atomicAdd(&lsum[col], s); atomicAdd(&lsq[col], s2); }
    }
    __syncthreads();
    if (tid < HID) { atomicAdd(&gsum[tid], lsum[tid]); atomicAdd(&gsq[tid], lsq[tid]); }
}

// ---------- output head, fused with last BN+residual: out = (sc*z+sh+y) @ Wout + bout ----------
__global__ __launch_bounds__(256)
void out_fused(const uint16_t* __restrict__ z, const uint16_t* __restrict__ y,
               const float* __restrict__ scp, const float* __restrict__ shp,
               const float* __restrict__ Wout, const float* __restrict__ bout,
               float* __restrict__ out) {
    __shared__ float wl[HID * DOUT];
    __shared__ float scl[HID], shl[HID];
    if (threadIdx.x < HID * DOUT) wl[threadIdx.x] = Wout[threadIdx.x];
    if (threadIdx.x < HID) { scl[threadIdx.x] = scp[threadIdx.x]; shl[threadIdx.x] = shp[threadIdx.x]; }
    __syncthreads();
    const int row = blockIdx.x * 256 + threadIdx.x;
    const uint4* zr = (const uint4*)(z + (size_t)row * HID);
    const uint4* yr = (const uint4*)(y + (size_t)row * HID);
    float a0 = bout[0], a1 = bout[1], a2 = bout[2];
#pragma unroll
    for (int g = 0; g < 8; ++g) {
        uint4 zq = zr[g];
        uint4 yq = yr[g];
        uint32_t zd[4] = {zq.x, zq.y, zq.z, zq.w};
        uint32_t yd[4] = {yq.x, yq.y, yq.z, yq.w};
#pragma unroll
        for (int c = 0; c < 4; ++c) {
            int f = g * 8 + 2 * c;
            float vl = scl[f]     * bf2f((uint16_t)(zd[c] & 0xffff)) + shl[f]     + bf2f((uint16_t)(yd[c] & 0xffff));
            float vh = scl[f + 1] * bf2f((uint16_t)(zd[c] >> 16))    + shl[f + 1] + bf2f((uint16_t)(yd[c] >> 16));
            a0 += vl * wl[f * 3 + 0] + vh * wl[(f + 1) * 3 + 0];
            a1 += vl * wl[f * 3 + 1] + vh * wl[(f + 1) * 3 + 1];
            a2 += vl * wl[f * 3 + 2] + vh * wl[(f + 1) * 3 + 2];
        }
    }
    float* o = out + (size_t)row * 3;
    o[0] = a0; o[1] = a1; o[2] = a2;
}

// ---------- host-side orchestration ----------
extern "C" void kernel_launch(void* const* d_in, const int* in_sizes, int n_in,
                              void* d_out, int out_size, void* d_ws, size_t ws_size,
                              hipStream_t stream) {
    (void)in_sizes; (void)n_in; (void)out_size; (void)ws_size;

    const float* x         = (const float*)d_in[0];
    const float* bn0g      = (const float*)d_in[1];
    const float* bn0b      = (const float*)d_in[2];
    const float* W0        = (const float*)d_in[3];
    const float* b0        = (const float*)d_in[4];
    const float* gamma0    = (const float*)d_in[5];
    const float* beta0     = (const float*)d_in[6];
    const float* W_hid     = (const float*)d_in[7];
    const float* b_hid     = (const float*)d_in[8];
    const float* gamma_hid = (const float*)d_in[9];
    const float* beta_hid  = (const float*)d_in[10];
    const float* Wout      = (const float*)d_in[11];
    const float* bout      = (const float*)d_in[12];
    float* out = (float*)d_out;

    char* ws = (char*)d_ws;
    const size_t actBytes = (size_t)BATCH * HID * 2;   // 128 MB bf16
    uint16_t* ybuf = (uint16_t*)(ws);
    uint16_t* zbuf = (uint16_t*)(ws + actBytes);
    uint16_t* wbf  = (uint16_t*)(ws + 2 * actBytes);   // transposed bf16 weights
    float* gsum  = (float*)(ws + 2 * actBytes + (size_t)NLAYERS * HID * HID * 2);
    float* gsq   = gsum + HID;
    float* scale = gsq + HID;
    float* shift = scale + HID;
    float* gsum0 = shift + HID;   // 4
    float* gsq0  = gsum0 + 4;     // 4
    float* sc0   = gsq0 + 4;      // 4
    float* sh0   = sc0 + 4;       // 4

    const float inv_n = 1.0f / (float)BATCH;

    // weights -> bf16, transposed (once per call, cheap)
    hipLaunchKernelGGL(conv_w_bf16_t, dim3(64), dim3(256), 0, stream, W_hid, wbf, NLAYERS);

    // bn0 over x
    hipMemsetAsync(gsum0, 0, 8 * sizeof(float), stream);
    hipLaunchKernelGGL(bn0_stats, dim3(1024), dim3(256), 0, stream,
                       (const float4*)x, gsum0, gsq0, (int)BATCH);
    hipLaunchKernelGGL(bn_finalize, dim3(1), dim3(64), 0, stream,
                       gsum0, gsq0, bn0g, bn0b, sc0, sh0, DIN, inv_n);

    // first hidden layer pre-activation: z0 = tanh(bn0(x)@W0+b0), + stats
    hipMemsetAsync(gsum, 0, 2 * HID * sizeof(float), stream);
    hipLaunchKernelGGL(layer0_kernel, dim3(BATCH / 256), dim3(256), 0, stream,
                       (const float4*)x, sc0, sh0, W0, b0, zbuf, gsum, gsq);
    hipLaunchKernelGGL(bn_finalize, dim3(1), dim3(64), 0, stream,
                       gsum, gsq, gamma0, beta0, scale, shift, HID, inv_n);

    // residual hidden blocks; previous BN affine fused into each GEMM's A staging
    for (int l = 0; l < NLAYERS; ++l) {
        hipMemsetAsync(gsum, 0, 2 * HID * sizeof(float), stream);
        hipLaunchKernelGGL(gemm64_fused, dim3(BATCH / 128), dim3(256), 0, stream,
                           zbuf, ybuf, scale, shift,
                           wbf + (size_t)l * HID * HID, b_hid + l * HID,
                           gsum, gsq, (l > 0) ? 1 : 0);
        hipLaunchKernelGGL(bn_finalize, dim3(1), dim3(64), 0, stream,
                           gsum, gsq, gamma_hid + l * HID, beta_hid + l * HID,
                           scale, shift, HID, inv_n);
    }

    // output head with the last BN + residual fused
    hipLaunchKernelGGL(out_fused, dim3(BATCH / 256), dim3(256), 0, stream,
                       zbuf, ybuf, scale, shift, Wout, bout, out);
}
